// RNNBasedEBM_74096775791361
// MI455X (gfx1250) — matrix operand order repaired and checked by reading, not compiled, not run on hardware
//
#include <hip/hip_runtime.h>

// ---------------------------------------------------------------------------
// GRU-based EBM:  B=256, S=512, I=512, H=1024
// Per-step fused GEMM [x_t | h] @ Wcat^T with bf16 WMMA, persistent kernel,
// one device barrier per timestep. h kept in bf16; h-phase A staging uses the
// CDNA5 async global->LDS path (global_load_async_to_lds_b128 / ASYNCcnt);
// x-phase converts f32->bf16 with single v_perm_b32 truncating packs.
// Wcat rows (N=4096): [wi_r|wh_r ; wi_z|wh_z ; wi_a|0 ; 0|wh_a]
// K loop split into x-phase (kc<16: tiles r,z,a_x) and h-phase (tiles r,z,a_h).
// ---------------------------------------------------------------------------

typedef __attribute__((ext_vector_type(16))) __bf16    v16bf;
typedef __attribute__((ext_vector_type(8)))  float     v8f;
typedef __attribute__((ext_vector_type(4)))  unsigned  v4u;
typedef __attribute__((ext_vector_type(4)))  float     v4f;

#define Bb   256
#define Ss   512
#define Ii   512
#define Hh   1024
#define Kk   1536          // I + H
#define Nn   4096          // 4*H (r, z, a_x, a_h)
#define KC   48            // Kk / 32
#define KCX  16            // x-part K chunks (512/32)
#define NWG  128           // 4 row-blocks x 32 j-blocks
#define NTHR 256

__device__ __forceinline__ unsigned short f2bf_rne(float f) {
    unsigned u = __float_as_uint(f);
    unsigned r = u + 0x7FFFu + ((u >> 16) & 1u);
    return (unsigned short)(r >> 16);
}

// pack two f32 -> two truncated bf16 in one v_perm_b32
__device__ __forceinline__ unsigned pack2bf(float lo, float hi) {
    return __builtin_amdgcn_perm(__float_as_uint(hi), __float_as_uint(lo), 0x07060302u);
}

__device__ __forceinline__ float bf2f(unsigned short s) {
    return __uint_as_float((unsigned)s << 16);
}

__device__ __forceinline__ void global_barrier(unsigned* cnt, unsigned target) {
    __threadfence();
    __syncthreads();
    if (threadIdx.x == 0) {
        __hip_atomic_fetch_add(cnt, 1u, __ATOMIC_RELEASE, __HIP_MEMORY_SCOPE_AGENT);
        while (__hip_atomic_load(cnt, __ATOMIC_ACQUIRE, __HIP_MEMORY_SCOPE_AGENT) < target) {
            __builtin_amdgcn_s_sleep(1);
        }
    }
    __syncthreads();
}

// ---------------------------------------------------------------------------
__global__ void init_state(unsigned short* __restrict__ hA, unsigned* __restrict__ cnt) {
    unsigned i = blockIdx.x * blockDim.x + threadIdx.x;
    if (i < Bb * Hh) hA[i] = 0;          // bf16 +0.0
    if (i == 0) *cnt = 0u;
}

// ---------------------------------------------------------------------------
// prepack Wcat (4096 x 1536) -> bf16 in per-lane WMMA B-fragment order (contiguous k, as A uses).
//   column n = ct*16 + (lane & 15);  K = kc*32 + 16*(lane>>4) + (2v+half)
// ---------------------------------------------------------------------------
__global__ void prepack_w(const float* __restrict__ wi, const float* __restrict__ wh,
                          unsigned short* __restrict__ Wbf) {
    unsigned e = blockIdx.x * blockDim.x + threadIdx.x;
    if (e >= (unsigned)(Nn * Kk)) return;
    unsigned b16  = e & 15u;
    unsigned lane = (e >> 4) & 31u;
    unsigned rest = e >> 9;
    unsigned kc   = rest % KC;
    unsigned ct   = rest / KC;
    unsigned n = ct * 16 + (lane & 15u);
    unsigned K = kc * 32 + ((lane >> 4) * 16) + b16;
    float v;
    if (n < 2048u)       v = (K < Ii) ? wi[(size_t)n * Ii + K] : wh[(size_t)n * Hh + (K - Ii)];
    else if (n < 3072u)  v = (K < Ii) ? wi[(size_t)n * Ii + K] : 0.f;                    // a_x
    else                 v = (K < Ii) ? 0.f : wh[(size_t)(n - 1024u) * Hh + (K - Ii)];   // a_h
    Wbf[e] = f2bf_rne(v);
}

// ---------------------------------------------------------------------------
// persistent GRU scan: one device barrier per timestep
// ---------------------------------------------------------------------------
__global__ void __launch_bounds__(NTHR, 1)
gru_scan(const float* __restrict__ x, const unsigned short* __restrict__ Wbf,
         const float* __restrict__ bi, const float* __restrict__ bn,
         unsigned short* __restrict__ hA, unsigned short* __restrict__ hB,
         unsigned* __restrict__ cnt) {
    const unsigned tid  = threadIdx.x;
    const unsigned wg   = blockIdx.x;
    const unsigned wgM  = wg & 3u;          // 64-row block
    const unsigned wgJ  = wg >> 2;          // 32-wide j block
    const unsigned wave = tid >> 5;
    const unsigned lane = tid & 31u;
    const unsigned rt   = wave & 3u;
    const unsigned jh   = wave >> 2;
    const unsigned m    = lane & 15u;
    const unsigned g    = lane >> 4;

    __shared__ unsigned short Atile[2][64 * 32];   // double-buffered bf16 A tile

    const unsigned srow = tid >> 2;             // staged row 0..63
    const unsigned sk0  = (tid & 3u) * 8;       // staged K offset
    const unsigned grow = wgM * 64 + srow;
    const float*          xrow  = x  + (size_t)grow * (Ss * Ii);
    const unsigned short* hrowA = hA + (size_t)grow * Hh;
    const unsigned short* hrowB = hB + (size_t)grow * Hh;

    // LDS byte offset of this thread's staging slot (Atile is the only LDS
    // object -> base offset 0); buffer p stride = 64*32*2 = 4096 bytes.
    const unsigned ldsSlot = (srow * 32u + sk0) * 2u;

    const v4u* Wq = (const v4u*)Wbf;
    const unsigned ct0  = wgJ * 2 + jh;         // r-gate column tile (0..63)
    const unsigned jcol = ct0 * 16 + m;
    const unsigned row0 = wgM * 64 + rt * 16 + g * 8;
    const unsigned arow = rt * 16 + m;

    const float bi_r = bi[jcol];
    const float bi_z = bi[1024 + jcol];
    const float bi_a = bi[2048 + jcol];
    const float bn_v = bn[jcol];

    for (int t = 0; t < Ss; ++t) {
        const unsigned short* hrow = (t & 1) ? hrowB : hrowA;
        const unsigned short* hc   = (t & 1) ? hB : hA;
        unsigned short*       hn   = (t & 1) ? hA : hB;

        v8f a0 = {}, a1 = {}, a2 = {}, a3 = {};

        // ---------------- x phase: kc = 0..15, tiles r, z, a_x --------------
        const float* xs = xrow + (size_t)t * Ii + sk0;
#pragma unroll 2
        for (int kc = 0; kc < KCX; ++kc) {
            const unsigned p = (unsigned)kc & 1u;
            {   // stage 8 x-elements: two b128 loads + v_perm packs + b128 LDS store
                v4f f0 = ((const v4f*)(xs + kc * 32))[0];
                v4f f1 = ((const v4f*)(xs + kc * 32))[1];
                v4u pk;
                pk.x = pack2bf(f0.x, f0.y);
                pk.y = pack2bf(f0.z, f0.w);
                pk.z = pack2bf(f1.x, f1.y);
                pk.w = pack2bf(f1.z, f1.w);
                *(v4u*)&Atile[p][srow * 32 + sk0] = pk;
            }
            __syncthreads();

            union AF { v4u q[2]; v16bf v; } af;
            af.q[0] = *(const v4u*)&Atile[p][arow * 32 +     16 * g];
            af.q[1] = *(const v4u*)&Atile[p][arow * 32 + 8 + 16 * g];

            union BF { v4u q[2]; v16bf v; } b0, b1, b2;
            size_t i0 = ((size_t)((ct0      ) * KC + kc) * 32 + lane) * 2;
            size_t i1 = ((size_t)((ct0 + 64 ) * KC + kc) * 32 + lane) * 2;
            size_t i2 = ((size_t)((ct0 + 128) * KC + kc) * 32 + lane) * 2;
            b0.q[0] = Wq[i0]; b0.q[1] = Wq[i0 + 1];
            b1.q[0] = Wq[i1]; b1.q[1] = Wq[i1 + 1];
            b2.q[0] = Wq[i2]; b2.q[1] = Wq[i2 + 1];
            a0 = __builtin_amdgcn_wmma_f32_16x16x32_bf16(false, af.v, false, b0.v, (short)0, a0, false, false);
            a1 = __builtin_amdgcn_wmma_f32_16x16x32_bf16(false, af.v, false, b1.v, (short)0, a1, false, false);
            a2 = __builtin_amdgcn_wmma_f32_16x16x32_bf16(false, af.v, false, b2.v, (short)0, a2, false, false);
        }

        // ---------------- h phase: kc = 16..47, tiles r, z, a_h -------------
        // staging is a pure bf16 copy -> use CDNA5 async global->LDS DMA path
        const unsigned short* hs = hrow + sk0;
#pragma unroll 2
        for (int kc = KCX; kc < KC; ++kc) {
            const unsigned p = (unsigned)kc & 1u;
            {
                unsigned lds_dst = p * 4096u + ldsSlot;
                unsigned long long gsrc =
                    (unsigned long long)(const void*)(hs + (size_t)(kc - KCX) * 32);
                asm volatile("global_load_async_to_lds_b128 %0, %1, off"
                             :: "v"(lds_dst), "v"(gsrc) : "memory");
                asm volatile("s_wait_asynccnt 0x0" ::: "memory");
            }
            __syncthreads();

            union AF { v4u q[2]; v16bf v; } af;
            af.q[0] = *(const v4u*)&Atile[p][arow * 32 +     16 * g];
            af.q[1] = *(const v4u*)&Atile[p][arow * 32 + 8 + 16 * g];

            union BF { v4u q[2]; v16bf v; } b0, b1, b2;
            size_t i0 = ((size_t)((ct0      ) * KC + kc) * 32 + lane) * 2;
            size_t i1 = ((size_t)((ct0 + 64 ) * KC + kc) * 32 + lane) * 2;
            size_t i3 = ((size_t)((ct0 + 192) * KC + kc) * 32 + lane) * 2;
            b0.q[0] = Wq[i0]; b0.q[1] = Wq[i0 + 1];
            b1.q[0] = Wq[i1]; b1.q[1] = Wq[i1 + 1];
            b2.q[0] = Wq[i3]; b2.q[1] = Wq[i3 + 1];
            a0 = __builtin_amdgcn_wmma_f32_16x16x32_bf16(false, af.v, false, b0.v, (short)0, a0, false, false);
            a1 = __builtin_amdgcn_wmma_f32_16x16x32_bf16(false, af.v, false, b1.v, (short)0, a1, false, false);
            a3 = __builtin_amdgcn_wmma_f32_16x16x32_bf16(false, af.v, false, b2.v, (short)0, a3, false, false);
        }

        // -------- gate phase: register-resident (a0=r, a1=z, a2=ia, a3=ha) --
#pragma unroll
        for (int v = 0; v < 8; ++v) {
            size_t hi = (size_t)(row0 + v) * Hh + jcol;
            float hv = bf2f(hc[hi]);
            float r  = 1.f / (1.f + __expf(-(a0[v] + bi_r)));
            float z  = 1.f / (1.f + __expf(-(a1[v] + bi_z)));
            float av = a2[v] + bi_a + r * (a3[v] + bn_v);
            av = fminf(fmaxf(av, -15.f), 15.f);
            float e2 = __expf(2.f * av);
            float n_ = (e2 - 1.f) / (e2 + 1.f);          // tanh
            hn[hi] = f2bf_rne(n_ + z * (hv - n_));
        }

        global_barrier(cnt, (unsigned)(t + 1) * NWG);
    }
}

// ---------------------------------------------------------------------------
// final linear: out[b] = h[b,:] . wl + bl   (final h in hA: S is even)
// ---------------------------------------------------------------------------
__global__ void final_linear(const unsigned short* __restrict__ h,
                             const float* __restrict__ wl,
                             const float* __restrict__ bl, float* __restrict__ out) {
    int b = blockIdx.x, tid = threadIdx.x;
    float s = 0.f;
    for (int j = tid; j < Hh; j += 256) s += bf2f(h[(size_t)b * Hh + j]) * wl[j];
    __shared__ float red[256];
    red[tid] = s;
    __syncthreads();
    for (int w = 128; w > 0; w >>= 1) {
        if (tid < w) red[tid] += red[tid + w];
        __syncthreads();
    }
    if (tid == 0) out[b] = red[0] + bl[0];
}

// ---------------------------------------------------------------------------
extern "C" void kernel_launch(void* const* d_in, const int* in_sizes, int n_in,
                              void* d_out, int out_size, void* d_ws, size_t ws_size,
                              hipStream_t stream) {
    const float* x  = (const float*)d_in[0];
    const float* wi = (const float*)d_in[1];
    const float* wh = (const float*)d_in[2];
    const float* bi = (const float*)d_in[3];
    const float* bn = (const float*)d_in[4];
    const float* wl = (const float*)d_in[5];
    const float* bl = (const float*)d_in[6];
    float* out = (float*)d_out;

    // workspace: [cnt 256B][hA bf16 512KB][hB bf16 512KB][Wbf bf16 12.6MB]
    char* ws = (char*)d_ws;
    unsigned*       cnt = (unsigned*)ws;
    unsigned short* hA  = (unsigned short*)(ws + 256);
    unsigned short* hB  = (unsigned short*)(ws + 256 + (size_t)Bb * Hh * 2);
    unsigned short* Wbf = (unsigned short*)(ws + 256 + 2 * (size_t)Bb * Hh * 2);

    init_state<<<(Bb * Hh + 255) / 256, 256, 0, stream>>>(hA, cnt);
    prepack_w<<<(Nn * Kk + 255) / 256, 256, 0, stream>>>(wi, wh, Wbf);
    gru_scan<<<NWG, NTHR, 0, stream>>>(x, Wbf, bi, bn, hA, hB, cnt);
    final_linear<<<Bb, 256, 0, stream>>>(hA, wl, bl, out);
}
